// SpatioTemporalAttention_29119878267072
// MI455X (gfx1250) — matrix operand order repaired and checked by reading, not compiled, not run on hardware
//
#include <hip/hip_runtime.h>
#include <hip/hip_bf16.h>
#include <math.h>

// ---------------------------------------------------------------------------
// SpatioTemporalAttention on gfx1250 (MI455X), bf16 WMMA pipeline, round 3.
//   N=16, C=192, T=120, V=100, HEADS=4, HEAD_CH=48
// 128-thread workgroups (4 wave32) compute a 64x64 output macro-tile.
// All edge guards on operand fills replaced by hoisted address clamps
// (garbage rows/cols are never stored); only the values-GEMM K-tail uses a
// branchless select-to-zero.  Scores kernel addresses simplified to k*V+u.
// ---------------------------------------------------------------------------

typedef __bf16 bf16_t;
typedef __attribute__((ext_vector_type(16))) __bf16 v16bf;
typedef __attribute__((ext_vector_type(8)))  __bf16 v8bf;
typedef __attribute__((ext_vector_type(8)))  float  v8f;

#define NB      16
#define CC      192
#define TT      120
#define VVD     100
#define TVD     12000            // T*V
#define NHEADS  4
#define HEAD_CH 48
#define KQK     (HEAD_CH * TT)   // 5760 contraction for scores
#define CTROWS  (CC * TT)        // 23040 rows of values GEMM
#define KD_OUT  (CC * NHEADS)    // 768 contraction for output proj

// --- WMMA fragment builders (ISA 7.12.2 layouts, 16-bit operands) ----------
__device__ __forceinline__ v16bf load_a_frag(const bf16_t* tile, int lane) {
  const int m  = lane & 15;
  const int hi = lane >> 4;
  v8bf lo = *reinterpret_cast<const v8bf*>(tile + m * 32 + 8 * hi);
  v8bf hv = *reinterpret_cast<const v8bf*>(tile + m * 32 + 16 + 8 * hi);
  return __builtin_shufflevector(lo, hv, 0, 1, 2, 3, 4, 5, 6, 7,
                                 8, 9, 10, 11, 12, 13, 14, 15);
}
__device__ __forceinline__ v16bf load_b_frag(const bf16_t* tile, int lane) {
  const int n  = lane & 15;
  const int hi = lane >> 4;
  v8bf lo = *reinterpret_cast<const v8bf*>(tile + n * 32 + 8 * hi);
  v8bf hv = *reinterpret_cast<const v8bf*>(tile + n * 32 + 16 + 8 * hi);
  return __builtin_shufflevector(lo, hv, 0, 1, 2, 3, 4, 5, 6, 7,
                                 8, 9, 10, 11, 12, 13, 14, 15);
}

#define WMMA_BF16(ACC, A, B)                                                   \
  ACC = __builtin_amdgcn_wmma_f32_16x16x32_bf16(false, (A), false, (B),        \
                                                (short)0, (ACC), false, false)

// ---------------------------------------------------------------------------
// Kernel 1: pointwise conv (channel GEMM)  Y[n,o,p] = sum_c W[o,c]*X[n,c,p]+b[o]
// X fp32 [NB][C][TV], Y bf16 same layout.  M=192, K=192, N=12000.
// grid = (188, 3, NB), block = 128.
// ---------------------------------------------------------------------------
__global__ __launch_bounds__(128)
void proj_bf16_kernel(const float* __restrict__ X, const float* __restrict__ W,
                      const float* __restrict__ bias, bf16_t* __restrict__ Y) {
  const int tid  = threadIdx.x;
  const int lane = tid & 31;
  const int wv   = tid >> 5;
  const int p0 = blockIdx.x * 64;
  const int m0 = blockIdx.y * 64;
  const int nb = blockIdx.z;

  __shared__ __align__(32) bf16_t As[64 * 32];
  __shared__ __align__(32) bf16_t Bs[64 * 32];

  const float* Xn = X + (size_t)nb * CC * TVD;
  // A-fill coords (hoisted)
  const int ar = tid >> 1, akb = (tid & 1) * 16;
  // B-fill coords: clamp column (garbage cols never stored)
  const int cn = tid & 63, bkb = (tid >> 6) * 16;
  const int pcl = (p0 + cn < TVD) ? (p0 + cn) : (TVD - 1);

  v8f acc[4] = {v8f{}, v8f{}, v8f{}, v8f{}};

  for (int k0 = 0; k0 < CC; k0 += 32) {
    {
      const float* src = W + (size_t)(m0 + ar) * CC + k0 + akb;
      v8bf h0, h1;
#pragma unroll
      for (int j = 0; j < 8; ++j) h0[j] = (bf16_t)src[j];
#pragma unroll
      for (int j = 0; j < 8; ++j) h1[j] = (bf16_t)src[8 + j];
      *reinterpret_cast<v8bf*>(&As[ar * 32 + akb]) = h0;
      *reinterpret_cast<v8bf*>(&As[ar * 32 + akb + 8]) = h1;
    }
    {
      const float* src = Xn + (size_t)(k0 + bkb) * TVD + pcl;
      v8bf h0, h1;
#pragma unroll
      for (int j = 0; j < 8; ++j) h0[j] = (bf16_t)src[(size_t)j * TVD];
#pragma unroll
      for (int j = 0; j < 8; ++j) h1[j] = (bf16_t)src[(size_t)(8 + j) * TVD];
      *reinterpret_cast<v8bf*>(&Bs[cn * 32 + bkb]) = h0;
      *reinterpret_cast<v8bf*>(&Bs[cn * 32 + bkb + 8]) = h1;
    }
    __syncthreads();
    v16bf a = load_a_frag(As + wv * 512, lane);
#pragma unroll
    for (int cb = 0; cb < 4; ++cb) {
      v16bf b = load_b_frag(Bs + cb * 512, lane);
      WMMA_BF16(acc[cb], a, b);
    }
    __syncthreads();
  }

  const int m = lane & 15, hi = lane >> 4;
#pragma unroll
  for (int cb = 0; cb < 4; ++cb) {
    const int p = p0 + cb * 16 + m;
    if (p < TVD) {
#pragma unroll
      for (int r = 0; r < 8; ++r) {
        const int row = m0 + wv * 16 + r + 8 * hi;
        Y[((size_t)nb * CC + row) * TVD + p] = (bf16_t)(acc[cb][r] + bias[row]);
      }
    }
  }
}

// ---------------------------------------------------------------------------
// Kernel 2: scores  S[n,s,u,v] = tanh(sum_{c,t} Q[..u]*K[..v] / 5760)*alpha + g
// Q layout [C][T][V] flat => element (c,t,u) is at k*V + u with k = c*T+t.
// Per (n,s): M=N=100 (pad 128), K=5760.  grid = (2, 2, NB*HEADS), block = 128.
// ---------------------------------------------------------------------------
__global__ __launch_bounds__(128)
void attn_kernel(const bf16_t* __restrict__ Q, const bf16_t* __restrict__ Kb,
                 const float* __restrict__ alphas, const float* __restrict__ gattn,
                 bf16_t* __restrict__ attn) {
  const int tid  = threadIdx.x;
  const int lane = tid & 31;
  const int wv   = tid >> 5;
  const int v0 = blockIdx.x * 64;
  const int u0 = blockIdx.y * 64;
  const int hs = blockIdx.z & (NHEADS - 1);
  const int nb = blockIdx.z >> 2;

  __shared__ __align__(32) bf16_t As[64 * 32];
  __shared__ __align__(32) bf16_t Bs[64 * 32];

  const bf16_t* Qh = Q + ((size_t)nb * CC + hs * HEAD_CH) * TVD;
  const bf16_t* Kh = Kb + ((size_t)nb * CC + hs * HEAD_CH) * TVD;

  // fill coords (hoisted): rows are u (A) / v (B); clamp (never stored if >=V)
  const int fr = tid & 63, fkb = (tid >> 6) * 16;
  const int ucl = (u0 + fr < VVD) ? (u0 + fr) : (VVD - 1);
  const int vcl = (v0 + fr < VVD) ? (v0 + fr) : (VVD - 1);

  v8f acc[4] = {v8f{}, v8f{}, v8f{}, v8f{}};

  for (int k0 = 0; k0 < KQK; k0 += 32) {
    {
      const bf16_t* src = Qh + (size_t)(k0 + fkb) * VVD + ucl;
      v8bf h0, h1;
#pragma unroll
      for (int j = 0; j < 8; ++j) h0[j] = src[(size_t)j * VVD];
#pragma unroll
      for (int j = 0; j < 8; ++j) h1[j] = src[(size_t)(8 + j) * VVD];
      *reinterpret_cast<v8bf*>(&As[fr * 32 + fkb]) = h0;
      *reinterpret_cast<v8bf*>(&As[fr * 32 + fkb + 8]) = h1;
    }
    {
      const bf16_t* src = Kh + (size_t)(k0 + fkb) * VVD + vcl;
      v8bf h0, h1;
#pragma unroll
      for (int j = 0; j < 8; ++j) h0[j] = src[(size_t)j * VVD];
#pragma unroll
      for (int j = 0; j < 8; ++j) h1[j] = src[(size_t)(8 + j) * VVD];
      *reinterpret_cast<v8bf*>(&Bs[fr * 32 + fkb]) = h0;
      *reinterpret_cast<v8bf*>(&Bs[fr * 32 + fkb + 8]) = h1;
    }
    __syncthreads();
    v16bf a = load_a_frag(As + wv * 512, lane);
#pragma unroll
    for (int cb = 0; cb < 4; ++cb) {
      v16bf b = load_b_frag(Bs + cb * 512, lane);
      WMMA_BF16(acc[cb], a, b);
    }
    __syncthreads();
  }

  const int m = lane & 15, hi = lane >> 4;
  const float alpha = alphas[hs];
  const float inv = 1.0f / (float)KQK;
#pragma unroll
  for (int cb = 0; cb < 4; ++cb) {
    const int v = v0 + cb * 16 + m;
    if (v < VVD) {
#pragma unroll
      for (int r = 0; r < 8; ++r) {
        const int u = u0 + wv * 16 + r + 8 * hi;
        if (u < VVD) {
          const float sv = tanhf(acc[cb][r] * inv) * alpha + gattn[u * VVD + v];
          attn[(((size_t)nb * NHEADS + hs) * VVD + u) * VVD + v] = (bf16_t)sv;
        }
      }
    }
  }
}

// ---------------------------------------------------------------------------
// Kernel 3 (per n): values[s, ct, v] = sum_u x_kv[ct, u] * attn[s, u, v]
// x_kv fp32 [CT][V]; attn bf16 [S][V][V]; vbuf bf16 [S*C][T][V].
// M=23040, K=100 (pad 128, K-tail select-to-zero), N=100 (pad 128, clamp).
// grid = (2, 360, 4), block = 128.
// ---------------------------------------------------------------------------
__global__ __launch_bounds__(128)
void values_kernel(const float* __restrict__ Xkv, const bf16_t* __restrict__ attn_n,
                   bf16_t* __restrict__ vbuf) {
  const int tid  = threadIdx.x;
  const int lane = tid & 31;
  const int wv   = tid >> 5;
  const int v0 = blockIdx.x * 64;
  const int row0 = blockIdx.y * 64;
  const int hs = blockIdx.z;

  __shared__ __align__(32) bf16_t As[64 * 32];
  __shared__ __align__(32) bf16_t Bs[64 * 32];

  const bf16_t* An = attn_n + (size_t)hs * VVD * VVD;
  const int ar = tid >> 1, akb = (tid & 1) * 16;
  const int cn = tid & 63, bkb = (tid >> 6) * 16;
  const int vcl = (v0 + cn < VVD) ? (v0 + cn) : (VVD - 1);

  v8f acc[4] = {v8f{}, v8f{}, v8f{}, v8f{}};

#pragma unroll
  for (int k0 = 0; k0 < 128; k0 += 32) {
    {  // A: contiguous in k (= u); K is the contraction -> select-to-zero tail
      const float* src = Xkv + (size_t)(row0 + ar) * VVD;
      v8bf h0, h1;
#pragma unroll
      for (int j = 0; j < 8; ++j) {
        const int k = k0 + akb + j;
        const float f = src[(k < VVD) ? k : (VVD - 1)];
        h0[j] = (k < VVD) ? (bf16_t)f : (bf16_t)0.0f;
      }
#pragma unroll
      for (int j = 0; j < 8; ++j) {
        const int k = k0 + akb + 8 + j;
        const float f = src[(k < VVD) ? k : (VVD - 1)];
        h1[j] = (k < VVD) ? (bf16_t)f : (bf16_t)0.0f;
      }
      *reinterpret_cast<v8bf*>(&As[ar * 32 + akb]) = h0;
      *reinterpret_cast<v8bf*>(&As[ar * 32 + akb + 8]) = h1;
    }
    {  // Bt: attn[k][v]; clamp v, select-to-zero K-tail
      v8bf h0, h1;
#pragma unroll
      for (int j = 0; j < 8; ++j) {
        const int k = k0 + bkb + j;
        const bf16_t f = An[(size_t)((k < VVD) ? k : (VVD - 1)) * VVD + vcl];
        h0[j] = (k < VVD) ? f : (bf16_t)0.0f;
      }
#pragma unroll
      for (int j = 0; j < 8; ++j) {
        const int k = k0 + bkb + 8 + j;
        const bf16_t f = An[(size_t)((k < VVD) ? k : (VVD - 1)) * VVD + vcl];
        h1[j] = (k < VVD) ? f : (bf16_t)0.0f;
      }
      *reinterpret_cast<v8bf*>(&Bs[cn * 32 + bkb]) = h0;
      *reinterpret_cast<v8bf*>(&Bs[cn * 32 + bkb + 8]) = h1;
    }
    __syncthreads();
    v16bf a = load_a_frag(As + wv * 512, lane);
#pragma unroll
    for (int cb = 0; cb < 4; ++cb) {
      v16bf b = load_b_frag(Bs + cb * 512, lane);
      WMMA_BF16(acc[cb], a, b);
    }
    __syncthreads();
  }

  const int m = lane & 15, hi = lane >> 4;
#pragma unroll
  for (int cb = 0; cb < 4; ++cb) {
    const int v = v0 + cb * 16 + m;
    if (v < VVD) {
#pragma unroll
      for (int r = 0; r < 8; ++r) {
        const int rg = row0 + wv * 16 + r + 8 * hi;  // < 23040 always
        vbuf[(size_t)hs * CTROWS * VVD + (size_t)rg * VVD + v] = (bf16_t)acc[cb][r];
      }
    }
  }
}

// ---------------------------------------------------------------------------
// Kernel 4 (per n): out[o, p] = sum_{sc<768} wp[o, sc] * vbuf[sc, p] + bp[o]
// vbuf bf16 [768][12000]; out fp32 [192][12000].  M=192, K=768, N=12000.
// grid = (188, 3), block = 128.
// ---------------------------------------------------------------------------
__global__ __launch_bounds__(128)
void outproj_kernel(const bf16_t* __restrict__ vbuf, const float* __restrict__ Wp,
                    const float* __restrict__ bp, float* __restrict__ Out) {
  const int tid  = threadIdx.x;
  const int lane = tid & 31;
  const int wv   = tid >> 5;
  const int p0 = blockIdx.x * 64;
  const int m0 = blockIdx.y * 64;

  __shared__ __align__(32) bf16_t As[64 * 32];
  __shared__ __align__(32) bf16_t Bs[64 * 32];

  const int ar = tid >> 1, akb = (tid & 1) * 16;
  const int cn = tid & 63, bkb = (tid >> 6) * 16;
  const int pcl = (p0 + cn < TVD) ? (p0 + cn) : (TVD - 1);

  v8f acc[4] = {v8f{}, v8f{}, v8f{}, v8f{}};

  for (int k0 = 0; k0 < KD_OUT; k0 += 32) {
    {
      const float* src = Wp + (size_t)(m0 + ar) * KD_OUT + k0 + akb;
      v8bf h0, h1;
#pragma unroll
      for (int j = 0; j < 8; ++j) h0[j] = (bf16_t)src[j];
#pragma unroll
      for (int j = 0; j < 8; ++j) h1[j] = (bf16_t)src[8 + j];
      *reinterpret_cast<v8bf*>(&As[ar * 32 + akb]) = h0;
      *reinterpret_cast<v8bf*>(&As[ar * 32 + akb + 8]) = h1;
    }
    {
      const bf16_t* src = vbuf + (size_t)(k0 + bkb) * TVD + pcl;
      v8bf h0, h1;
#pragma unroll
      for (int j = 0; j < 8; ++j) h0[j] = src[(size_t)j * TVD];
#pragma unroll
      for (int j = 0; j < 8; ++j) h1[j] = src[(size_t)(8 + j) * TVD];
      *reinterpret_cast<v8bf*>(&Bs[cn * 32 + bkb]) = h0;
      *reinterpret_cast<v8bf*>(&Bs[cn * 32 + bkb + 8]) = h1;
    }
    __syncthreads();
    v16bf a = load_a_frag(As + wv * 512, lane);
#pragma unroll
    for (int cb = 0; cb < 4; ++cb) {
      v16bf b = load_b_frag(Bs + cb * 512, lane);
      WMMA_BF16(acc[cb], a, b);
    }
    __syncthreads();
  }

  const int m = lane & 15, hi = lane >> 4;
#pragma unroll
  for (int cb = 0; cb < 4; ++cb) {
    const int p = p0 + cb * 16 + m;
    if (p < TVD) {
#pragma unroll
      for (int r = 0; r < 8; ++r) {
        const int row = m0 + wv * 16 + r + 8 * hi;
        Out[(size_t)row * TVD + p] = acc[cb][r] + bp[row];
      }
    }
  }
}

// ---------------------------------------------------------------------------
// Launch: Q proj, K proj, scores, then per-n {values, out-proj} reusing one
// 18.4MB bf16 chunk.  Workspace use: Q(73.7MB)+K(73.7MB)+attn(1.3MB)+V(18.4MB).
// ---------------------------------------------------------------------------
extern "C" void kernel_launch(void* const* d_in, const int* in_sizes, int n_in,
                              void* d_out, int out_size, void* d_ws, size_t ws_size,
                              hipStream_t stream) {
  const float* x_q    = (const float*)d_in[0];
  const float* x_kv   = (const float*)d_in[1];
  const float* wq     = (const float*)d_in[2];
  const float* bq     = (const float*)d_in[3];
  const float* wk     = (const float*)d_in[4];
  const float* bk     = (const float*)d_in[5];
  const float* wp     = (const float*)d_in[6];
  const float* bp     = (const float*)d_in[7];
  const float* alphas = (const float*)d_in[8];
  const float* gattn  = (const float*)d_in[9];
  float* out = (float*)d_out;

  bf16_t* Qb = (bf16_t*)d_ws;                              // [16][192][12000]
  bf16_t* Kb = Qb + (size_t)NB * CC * TVD;                 // [16][192][12000]
  bf16_t* Ab = Kb + (size_t)NB * CC * TVD;                 // [16][4][100][100]
  bf16_t* Vb = Ab + (size_t)NB * NHEADS * VVD * VVD;       // [768][12000] (per-n reuse)

  const dim3 gProj((TVD + 63) / 64, CC / 64, NB);          // (188, 3, 16)
  proj_bf16_kernel<<<gProj, 128, 0, stream>>>(x_q,  wq, bq, Qb);
  proj_bf16_kernel<<<gProj, 128, 0, stream>>>(x_kv, wk, bk, Kb);

  attn_kernel<<<dim3(2, 2, NB * NHEADS), 128, 0, stream>>>(Qb, Kb, alphas, gattn, Ab);

  for (int n = 0; n < NB; ++n) {
    values_kernel<<<dim3(2, CTROWS / 64, NHEADS), 128, 0, stream>>>(
        x_kv + (size_t)n * CC * TVD, Ab + (size_t)n * NHEADS * VVD * VVD, Vb);
    outproj_kernel<<<dim3((TVD + 63) / 64, CC / 64), 128, 0, stream>>>(
        Vb, wp, bp, out + (size_t)n * CC * TVD);
  }
}
